// PathPredictor_79087527789099
// MI455X (gfx1250) — matrix operand order repaired and checked by reading, not compiled, not run on hardware
//
#include <hip/hip_runtime.h>
#include <cstdint>
#include <cstddef>

// ---------------- problem constants ----------------
static constexpr int E    = 200;     // embedding dim
static constexpr int KP   = 224;     // E padded to multiple of 32
static constexpr int E2   = 400;     // 2E
static constexpr int KP2  = 416;     // 2E padded to multiple of 32
static constexpr int RR   = 480;     // num relations (already mult of 32)
static constexpr int TT   = 8;       // time steps
static constexpr int NENT = 40000;   // entities
static constexpr int BB   = 1024;    // batch
static constexpr int G3   = 600;     // 3E (GRU gates)
static constexpr int G3P  = 608;     // padded to mult of 16
static constexpr int EP16 = 208;     // 200 padded to mult of 16 (N-dim pad)
static constexpr float NEG_INF_ = -1000000000.0f;

typedef __attribute__((ext_vector_type(16))) __bf16 v16bf;
typedef __attribute__((ext_vector_type(8)))  __bf16 v8bf;
typedef __attribute__((ext_vector_type(8)))  float  v8f;

__device__ __forceinline__ uint16_t f2bf(float x) {
  uint32_t u = __float_as_uint(x);
  u += 0x7FFFu + ((u >> 16) & 1u);       // round-to-nearest-even
  return (uint16_t)(u >> 16);
}
__device__ __forceinline__ float sigmoidf_(float x) { return 1.0f / (1.0f + expf(-x)); }
__device__ __forceinline__ float softplusf_(float x) {
  return fmaxf(x, 0.0f) + log1pf(expf(-fabsf(x)));
}

// -------- WMMA fragment builders (bf16, 16x16x32) --------
// A (MxK): lanes 0-15 -> M=lane, K: [k..k+7] then [k+16..k+23]
//          lanes 16-31 -> M=lane-16, K: [k+8..k+15] then [k+24..k+31]
// B (KxN): lanes 0-15 -> N=lane,  K: [k..k+7] then [k+16..k+23]
//          lanes 16-31 -> N=lane-16, K: [k+8..k+15] then [k+24..k+31]
__device__ __forceinline__ v16bf frag_cat(v8bf lo, v8bf hi) {
  return __builtin_shufflevector(lo, hi, 0,1,2,3,4,5,6,7,8,9,10,11,12,13,14,15);
}

// ---------------- generic bf16 WMMA GEMM:  C = A(MxK) * B(NxK)^T + bias ----------------
// A: bf16 raw, lda = padded K (mult of 32). B: bf16 raw, ldb = padded K.
// Optional f32 output C (ldc) and bf16 mirror Cbf (ldcbf, pad cols pre-zeroed).
__global__ __launch_bounds__(256) void gemm_bf16_kernel(
    const uint16_t* __restrict__ A, int lda,
    const uint16_t* __restrict__ Bm, int ldb,
    const float* __restrict__ bias,
    float* __restrict__ C, int ldc,
    uint16_t* __restrict__ Cbf, int ldcbf,
    int M, int N, int Np, int Kp)
{
  const int wave = threadIdx.x >> 5;
  const int lane = threadIdx.x & 31;
  const int m0 = blockIdx.y * 32 + (wave >> 2) * 16;
  const int n0 = blockIdx.x * 64 + (wave & 3) * 16;
  if (m0 >= M || n0 >= Np) return;

  const int mi = lane & 15;
  const int hh = lane >> 4;                 // half-wave selector
  const __bf16* ap = (const __bf16*)A + (size_t)(m0 + mi) * lda + hh * 8;
  const __bf16* bp = (const __bf16*)Bm + (size_t)(n0 + mi) * ldb + hh * 8;

  v8f acc = {0.f,0.f,0.f,0.f,0.f,0.f,0.f,0.f};
  for (int k = 0; k < Kp; k += 32) {
    v8bf alo = *(const v8bf*)(ap + k);
    v8bf ahi = *(const v8bf*)(ap + k + 16);
    v8bf blo = *(const v8bf*)(bp + k);
    v8bf bhi = *(const v8bf*)(bp + k + 16);
    v16bf af = frag_cat(alo, ahi);
    v16bf bf = frag_cat(blo, bhi);
    acc = __builtin_amdgcn_wmma_f32_16x16x32_bf16(false, af, false, bf,
                                                  (short)0, acc, false, false);
  }

  const int col = n0 + mi;
  if (col < N) {
    const float bv = bias ? bias[col] : 0.0f;
#pragma unroll
    for (int r = 0; r < 8; ++r) {
      const int row = m0 + r + hh * 8;
      if (row < M) {
        const float v = acc[r] + bv;
        if (C)   C[(size_t)row * ldc + col] = v;
        if (Cbf) Cbf[(size_t)row * ldcbf + col] = f2bf(v);
      }
    }
  }
}

// ---------------- big fused GEMM: score = sigmoid(AB @ P^T), + path-loss reduction ------
// M=BB(1024), N=NENT(40000), K=KP(224). Exact tiling: grid (625, 32), no early exit.
// A/B panels staged in LDS with GLOBAL_LOAD_ASYNC_TO_LDS_B128 (ASYNCcnt-tracked),
// then WMMA fragments are fed from LDS (ds_load_b128).
__global__ __launch_bounds__(256) void score_kernel(
    const uint16_t* __restrict__ A,     // AB_bf   (BB x KP)
    const uint16_t* __restrict__ Bm,    // P_bf    (NENT x KP)
    const int* __restrict__ triples,    // (BB x 3) int32
    float* __restrict__ outScore,       // (BB x NENT)
    float* __restrict__ accum)          // accum[0] += sum of BCE elements
{
  __shared__ __align__(16) uint16_t sA[32 * KP];   // 14336 B
  __shared__ __align__(16) uint16_t sB[64 * KP];   // 28672 B
  __shared__ float red[256];

  const int tid  = threadIdx.x;
  const int mblk = blockIdx.y * 32;
  const int nblk = blockIdx.x * 64;

  // ---- async stage: A panel (32 rows) + B panel (64 rows), both KP bf16 wide ----
  constexpr int CHA = 32 * KP * 2 / 16;   // 896  16-byte chunks
  constexpr int CHB = 64 * KP * 2 / 16;   // 1792 16-byte chunks
  {
    const char* gA = (const char*)(A  + (size_t)mblk * KP);
    const char* gB = (const char*)(Bm + (size_t)nblk * KP);
    for (int c = tid; c < CHA + CHB; c += 256) {
      uint64_t gaddr;
      uint32_t laddr;
      if (c < CHA) {
        gaddr = (uint64_t)(uintptr_t)(gA + (size_t)c * 16);
        laddr = (uint32_t)(uintptr_t)((const char*)sA + (size_t)c * 16);
      } else {
        const int c2 = c - CHA;
        gaddr = (uint64_t)(uintptr_t)(gB + (size_t)c2 * 16);
        laddr = (uint32_t)(uintptr_t)((const char*)sB + (size_t)c2 * 16);
      }
      asm volatile("global_load_async_to_lds_b128 %0, %1, off"
                   :: "v"(laddr), "v"(gaddr) : "memory");
    }
    asm volatile("s_wait_asynccnt 0x0" ::: "memory");
  }
  __syncthreads();

  const int wave = tid >> 5;
  const int lane = tid & 31;
  const int mi = lane & 15;
  const int hh = lane >> 4;
  const int mw = (wave >> 2) * 16;        // wave M offset within block tile
  const int nw = (wave & 3) * 16;         // wave N offset within block tile

  const __bf16* ap = (const __bf16*)sA + (size_t)(mw + mi) * KP + hh * 8;
  const __bf16* bp = (const __bf16*)sB + (size_t)(nw + mi) * KP + hh * 8;

  v8f acc = {0.f,0.f,0.f,0.f,0.f,0.f,0.f,0.f};
#pragma unroll
  for (int k = 0; k < KP; k += 32) {
    v16bf af = frag_cat(*(const v8bf*)(ap + k), *(const v8bf*)(ap + k + 16));
    v16bf bf = frag_cat(*(const v8bf*)(bp + k), *(const v8bf*)(bp + k + 16));
    acc = __builtin_amdgcn_wmma_f32_16x16x32_bf16(false, af, false, bf,
                                                  (short)0, acc, false, false);
  }

  const int col = nblk + nw + mi;
  float lsum = 0.0f;
#pragma unroll
  for (int r = 0; r < 8; ++r) {
    const int row = mblk + mw + r + hh * 8;
    const float l = acc[r];
    outScore[(size_t)row * NENT + col] = sigmoidf_(l);
    const int obj = triples[row * 3 + 2];
    const float x = (col == obj) ? -l : l;   // label=1 -> softplus(-l), else softplus(l)
    lsum += softplusf_(x);
  }

  red[tid] = lsum;
  __syncthreads();
  for (int s = 128; s > 0; s >>= 1) {
    if (tid < s) red[tid] += red[tid + s];
    __syncthreads();
  }
  if (tid == 0) atomicAdd(accum, red[0]);
}

// ---------------- masked softmax over relations (per batch row) ----------------
__global__ __launch_bounds__(256) void softmax_kernel(
    const float* __restrict__ pemb,   // (BB, TT*RR)
    const float* __restrict__ base,   // (BB, RR)
    uint16_t* __restrict__ a_bf,      // (BB, RR) bf16 out
    int t)
{
  __shared__ float red[256];
  const int row = blockIdx.x;
  const float* p = pemb + (size_t)row * (TT * RR) + (size_t)t * RR;
  const float* b = base + (size_t)row * RR;

  float vals[2];
  float vmax = -INFINITY;
#pragma unroll
  for (int i = 0; i < 2; ++i) {
    const int c = threadIdx.x + i * 256;
    float v = -INFINITY;
    if (c < RR) {
      v = p[c] * b[c];
      if (v == 0.0f) v = NEG_INF_;
    }
    vals[i] = v;
    vmax = fmaxf(vmax, v);
  }
  red[threadIdx.x] = vmax;
  __syncthreads();
  for (int s = 128; s > 0; s >>= 1) {
    if (threadIdx.x < s) red[threadIdx.x] = fmaxf(red[threadIdx.x], red[threadIdx.x + s]);
    __syncthreads();
  }
  const float m = red[0];
  __syncthreads();

  float ssum = 0.0f;
#pragma unroll
  for (int i = 0; i < 2; ++i) {
    const int c = threadIdx.x + i * 256;
    float e = (c < RR) ? expf(vals[i] - m) : 0.0f;
    vals[i] = e;
    ssum += e;
  }
  red[threadIdx.x] = ssum;
  __syncthreads();
  for (int s = 128; s > 0; s >>= 1) {
    if (threadIdx.x < s) red[threadIdx.x] += red[threadIdx.x + s];
    __syncthreads();
  }
  const float inv = 1.0f / red[0];
#pragma unroll
  for (int i = 0; i < 2; ++i) {
    const int c = threadIdx.x + i * 256;
    if (c < RR) a_bf[(size_t)row * RR + c] = f2bf(vals[i] * inv);
  }
}

// ---------------- GRU gate combine ----------------
__global__ void gru_gate_kernel(const float* __restrict__ gi,
                                const float* __restrict__ gh,
                                const float* __restrict__ hprev,
                                float* __restrict__ hout,
                                uint16_t* __restrict__ hbf)
{
  const int idx = blockIdx.x * blockDim.x + threadIdx.x;
  if (idx >= BB * E) return;
  const int row = idx / E, col = idx % E;
  const size_t b3 = (size_t)row * G3 + col;
  const float r = sigmoidf_(gi[b3] + gh[b3]);
  const float z = sigmoidf_(gi[b3 + E] + gh[b3 + E]);
  const float n = tanhf(gi[b3 + 2 * E] + r * gh[b3 + 2 * E]);
  const float h = (1.0f - z) * n + z * hprev[idx];
  hout[idx] = h;
  if (hbf) hbf[(size_t)row * KP + col] = f2bf(h);
}

// ---------------- fp32 -> bf16 convert with row/col zero-padding ----------------
__global__ void convert_pad_kernel(const float* __restrict__ src, uint16_t* __restrict__ dst,
                                   int M, int Mp, int K, int Kp)
{
  const size_t idx = (size_t)blockIdx.x * blockDim.x + threadIdx.x;
  if (idx >= (size_t)Mp * Kp) return;
  const int r = (int)(idx / Kp), c = (int)(idx % Kp);
  const float v = (r < M && c < K) ? src[(size_t)r * K + c] : 0.0f;
  dst[idx] = f2bf(v);
}

// ---------------- mapped_rel^T (bf16): dst[n*RR+m] = mapped_rel[m*E+n] ----------------
__global__ void transpose_bf_kernel(const float* __restrict__ src, uint16_t* __restrict__ dst)
{
  const int idx = blockIdx.x * blockDim.x + threadIdx.x;
  if (idx >= E * RR) return;
  const int n = idx / RR, m = idx % RR;
  dst[(size_t)n * RR + m] = f2bf(src[(size_t)m * E + n]);
}

// ---------------- gather q_rel = mapped_rel[rel] (f32 + padded bf16) ----------------
__global__ void gather_qrel_kernel(const float* __restrict__ mapped,
                                   const int* __restrict__ triples,
                                   float* __restrict__ qf, uint16_t* __restrict__ qbf)
{
  const int idx = blockIdx.x * blockDim.x + threadIdx.x;
  if (idx >= BB * KP) return;
  const int row = idx / KP, c = idx % KP;
  const int rel = triples[row * 3 + 1];
  if (c < E) {
    const float v = mapped[(size_t)rel * E + c];
    qf[(size_t)row * E + c] = v;
    qbf[idx] = f2bf(v);
  } else {
    qbf[idx] = 0;
  }
}

// ---------------- gather sub_emb = pre_emb[sub] (padded bf16) ----------------
__global__ void gather_sub_kernel(const float* __restrict__ pre,
                                  const int* __restrict__ triples,
                                  uint16_t* __restrict__ sbf)
{
  const int idx = blockIdx.x * blockDim.x + threadIdx.x;
  if (idx >= BB * KP) return;
  const int row = idx / KP, c = idx % KP;
  sbf[idx] = (c < E) ? f2bf(pre[(size_t)triples[row * 3 + 0] * E + c]) : (uint16_t)0;
}

// ---------------- predicted_hist = 0.1*t3 + q_rel (f32 + bf16) ----------------
__global__ void ph_kernel(const float* __restrict__ t3, const float* __restrict__ qf,
                          float* __restrict__ phf, uint16_t* __restrict__ phbf)
{
  const int idx = blockIdx.x * blockDim.x + threadIdx.x;
  if (idx >= BB * E) return;
  const int row = idx / E, col = idx % E;
  const float v = 0.1f * t3[idx] + qf[idx];
  phf[idx] = v;
  phbf[(size_t)row * KP + col] = f2bf(v);
}

// ---------------- AB = SA * gru_hidden2 (elementwise, padded bf16) ----------------
__global__ void ab_kernel(const float* __restrict__ sa, const float* __restrict__ g2,
                          uint16_t* __restrict__ abf)
{
  const int idx = blockIdx.x * blockDim.x + threadIdx.x;
  if (idx >= BB * KP) return;
  const int row = idx / KP, c = idx % KP;
  abf[idx] = (c < E) ? f2bf(sa[(size_t)row * E + c] * g2[(size_t)row * E + c]) : (uint16_t)0;
}

// ---------------- match loss: sum((a-b)^2) ----------------
__global__ void mse_kernel(const float* __restrict__ a, const float* __restrict__ b,
                           float* __restrict__ accum, int n)
{
  __shared__ float red[256];
  const int idx = blockIdx.x * blockDim.x + threadIdx.x;
  float d = 0.0f;
  if (idx < n) { const float t = a[idx] - b[idx]; d = t * t; }
  red[threadIdx.x] = d;
  __syncthreads();
  for (int s = 128; s > 0; s >>= 1) {
    if (threadIdx.x < s) red[threadIdx.x] += red[threadIdx.x + s];
    __syncthreads();
  }
  if (threadIdx.x == 0) atomicAdd(accum, red[0]);
}

__global__ void zero_kernel(uint32_t* __restrict__ p, size_t n)
{
  size_t i = (size_t)blockIdx.x * blockDim.x + threadIdx.x;
  const size_t stride = (size_t)gridDim.x * blockDim.x;
  for (; i < n; i += stride) p[i] = 0;
}

__global__ void finalize_kernel(const float* __restrict__ accum, float* __restrict__ out)
{
  out[0] = accum[0] / (float)(BB * E);
  out[1] = accum[1] / ((float)BB * (float)NENT);
}

// =========================== host orchestration ===========================
extern "C" void kernel_launch(void* const* d_in, const int* in_sizes, int n_in,
                              void* d_out, int out_size, void* d_ws, size_t ws_size,
                              hipStream_t stream)
{
  (void)in_sizes; (void)n_in; (void)out_size; (void)ws_size;

  const float* pre_emb = (const float*)d_in[0];
  const float* r_emb   = (const float*)d_in[1];
  const float* pemb    = (const float*)d_in[2];
  const int*   triples = (const int*)d_in[3];
  const float* W1  = (const float*)d_in[6];  const float* b1  = (const float*)d_in[7];
  const float* W2  = (const float*)d_in[8];  const float* b2  = (const float*)d_in[9];
  const float* Wa  = (const float*)d_in[10]; const float* ba  = (const float*)d_in[11];
  const float* Wh1 = (const float*)d_in[12]; const float* bh1 = (const float*)d_in[13];
  const float* Wh2 = (const float*)d_in[14]; const float* bh2 = (const float*)d_in[15];
  const float* Wal = (const float*)d_in[16]; const float* bal = (const float*)d_in[17];
  const float* wih = (const float*)d_in[18]; const float* whh = (const float*)d_in[19];
  const float* bih = (const float*)d_in[20]; const float* bhh = (const float*)d_in[21];

  float* out   = (float*)d_out;
  float* score = out + 2;

  // ---- workspace bump allocator (256B aligned) ----
  char* w = (char*)d_ws;
  size_t off = 0;
  auto alloc = [&](size_t bytes) -> void* {
    void* p = w + off;
    off += (bytes + 255) & ~(size_t)255;
    return p;
  };
  float*    accum   = (float*)alloc(2 * sizeof(float));
  float*    t1      = (float*)alloc((size_t)RR * E2 * 4);
  float*    mr      = (float*)alloc((size_t)RR * E * 4);        // mapped_rel f32
  float*    qf      = (float*)alloc((size_t)BB * E * 4);        // q_rel f32
  float*    base    = (float*)alloc((size_t)BB * RR * 4);
  float*    gi      = (float*)alloc((size_t)BB * G3 * 4);
  float*    gh      = (float*)alloc((size_t)BB * G3 * 4);
  float*    hbuf    = (float*)alloc((size_t)BB * E * 4);        // GRU hidden f32
  float*    t3      = (float*)alloc((size_t)BB * E * 4);
  float*    ph      = (float*)alloc((size_t)BB * E * 4);        // predicted_hist f32
  float*    g2      = (float*)alloc((size_t)BB * E * 4);        // gru_hidden2 f32
  float*    sa      = (float*)alloc((size_t)BB * E * 4);        // lin(sub_emb,Wal,bal)

  uint16_t* remb_bf = (uint16_t*)alloc((size_t)RR * KP * 2);
  uint16_t* W1_bf   = (uint16_t*)alloc((size_t)E2 * KP * 2);
  uint16_t* t1_bf   = (uint16_t*)alloc((size_t)RR * KP2 * 2);
  uint16_t* W2_bf   = (uint16_t*)alloc((size_t)EP16 * KP2 * 2);
  uint16_t* mr_bf   = (uint16_t*)alloc((size_t)RR * KP * 2);
  uint16_t* mrT_bf  = (uint16_t*)alloc((size_t)EP16 * RR * 2);
  uint16_t* q_bf    = (uint16_t*)alloc((size_t)BB * KP * 2);
  uint16_t* Wa_bf   = (uint16_t*)alloc((size_t)EP16 * KP * 2);
  uint16_t* qa_bf   = (uint16_t*)alloc((size_t)BB * KP * 2);
  uint16_t* a_bf    = (uint16_t*)alloc((size_t)BB * RR * 2);
  uint16_t* x_bf    = (uint16_t*)alloc((size_t)BB * KP * 2);
  uint16_t* wih_bf  = (uint16_t*)alloc((size_t)G3P * KP * 2);
  uint16_t* whh_bf  = (uint16_t*)alloc((size_t)G3P * KP * 2);
  uint16_t* h_bf    = (uint16_t*)alloc((size_t)BB * KP * 2);
  uint16_t* Wh1_bf  = (uint16_t*)alloc((size_t)EP16 * KP * 2);
  uint16_t* Wh2_bf  = (uint16_t*)alloc((size_t)EP16 * KP * 2);
  uint16_t* t2_bf   = (uint16_t*)alloc((size_t)BB * KP * 2);
  uint16_t* ph_bf   = (uint16_t*)alloc((size_t)BB * KP * 2);
  uint16_t* Wal_bf  = (uint16_t*)alloc((size_t)EP16 * KP * 2);
  uint16_t* sub_bf  = (uint16_t*)alloc((size_t)BB * KP * 2);
  uint16_t* AB_bf   = (uint16_t*)alloc((size_t)BB * KP * 2);
  uint16_t* pre_bf  = (uint16_t*)alloc((size_t)NENT * KP * 2);
  uint16_t* P_bf    = (uint16_t*)alloc((size_t)NENT * KP * 2);

  // ---- 0) zero everything we use (pads, h0, accumulators) ----
  {
    const size_t nwords = (off + 3) / 4;
    int blocks = (int)((nwords + 255) / 256);
    if (blocks > 32768) blocks = 32768;
    zero_kernel<<<blocks, 256, 0, stream>>>((uint32_t*)d_ws, nwords);
  }

  auto conv = [&](const float* src, uint16_t* dst, int M, int Mp, int K, int Kp) {
    const size_t n = (size_t)Mp * Kp;
    convert_pad_kernel<<<(int)((n + 255) / 256), 256, 0, stream>>>(src, dst, M, Mp, K, Kp);
  };
  auto gemm = [&](const uint16_t* A, int lda, const uint16_t* Bm, int ldb,
                  const float* bias, float* C, int ldc, uint16_t* Cbf, int ldcbf,
                  int M, int N, int Np, int Kp) {
    dim3 grid((Np + 63) / 64, (M + 31) / 32);
    gemm_bf16_kernel<<<grid, 256, 0, stream>>>(A, lda, Bm, ldb, bias, C, ldc,
                                               Cbf, ldcbf, M, N, Np, Kp);
  };

  // ---- 1) convert weights / inputs to padded bf16 ----
  conv(r_emb, remb_bf, RR, RR, E, KP);
  conv(W1,  W1_bf,  E2, E2, E, KP);
  conv(W2,  W2_bf,  E,  EP16, E2, KP2);
  conv(Wa,  Wa_bf,  E,  EP16, E, KP);
  conv(Wh1, Wh1_bf, E,  EP16, E, KP);
  conv(Wh2, Wh2_bf, E,  EP16, E, KP);
  conv(Wal, Wal_bf, E,  EP16, E, KP);
  conv(wih, wih_bf, G3, G3P, E, KP);
  conv(whh, whh_bf, G3, G3P, E, KP);
  conv(pre_emb, pre_bf, NENT, NENT, E, KP);

  // ---- 2) mapped_rel = lin(lin(r_emb,W1,b1),W2,b2) ----
  gemm(remb_bf, KP, W1_bf, KP, b1, t1, E2, t1_bf, KP2, RR, E2, E2, KP);
  gemm(t1_bf, KP2, W2_bf, KP2, b2, mr, E, mr_bf, KP, RR, E, EP16, KP2);
  transpose_bf_kernel<<<(E * RR + 255) / 256, 256, 0, stream>>>(mr, mrT_bf);

  // ---- 3) q_rel gather; base = lin(q_rel,Wa,ba) @ mapped_rel^T ----
  gather_qrel_kernel<<<(BB * KP + 255) / 256, 256, 0, stream>>>(mr, triples, qf, q_bf);
  gemm(q_bf, KP, Wa_bf, KP, ba, nullptr, 0, qa_bf, KP, BB, E, EP16, KP);
  gemm(qa_bf, KP, mr_bf, KP, nullptr, base, RR, nullptr, 0, BB, RR, RR, KP);

  // ---- 4) GRU scan over T steps ----
  for (int t = 0; t < TT; ++t) {
    softmax_kernel<<<BB, 256, 0, stream>>>(pemb, base, a_bf, t);
    gemm(a_bf, RR, mrT_bf, RR, nullptr, nullptr, 0, x_bf, KP, BB, E, EP16, RR); // x = a @ mapped_rel
    gemm(x_bf, KP, wih_bf, KP, bih, gi, G3, nullptr, 0, BB, G3, G3P, KP);
    gemm(h_bf, KP, whh_bf, KP, bhh, gh, G3, nullptr, 0, BB, G3, G3P, KP);
    gru_gate_kernel<<<(BB * E + 255) / 256, 256, 0, stream>>>(gi, gh, hbuf, hbuf, h_bf);
  }

  // ---- 5) predicted_hist = 0.1*lin(lin(q_rel,Wh1,bh1),Wh2,bh2) + q_rel ----
  gemm(q_bf, KP, Wh1_bf, KP, bh1, nullptr, 0, t2_bf, KP, BB, E, EP16, KP);
  gemm(t2_bf, KP, Wh2_bf, KP, bh2, t3, E, nullptr, 0, BB, E, EP16, KP);
  ph_kernel<<<(BB * E + 255) / 256, 256, 0, stream>>>(t3, qf, ph, ph_bf);

  // ---- 6) match_loss = mean((predicted_hist - gru_hidden)^2) ----
  mse_kernel<<<(BB * E + 255) / 256, 256, 0, stream>>>(ph, hbuf, accum, BB * E);

  // ---- 7) gru_hidden2 = gru_cell(q_rel, predicted_hist) ----
  gemm(q_bf, KP, wih_bf, KP, bih, gi, G3, nullptr, 0, BB, G3, G3P, KP);
  gemm(ph_bf, KP, whh_bf, KP, bhh, gh, G3, nullptr, 0, BB, G3, G3P, KP);
  gru_gate_kernel<<<(BB * E + 255) / 256, 256, 0, stream>>>(gi, gh, ph, g2, nullptr);

  // ---- 8) scoring: AB = lin(sub_emb,Wal,bal) * g2 ; P = lin(pre_emb,Wal,bal) ----
  gather_sub_kernel<<<(BB * KP + 255) / 256, 256, 0, stream>>>(pre_emb, triples, sub_bf);
  gemm(sub_bf, KP, Wal_bf, KP, bal, sa, E, nullptr, 0, BB, E, EP16, KP);
  ab_kernel<<<(BB * KP + 255) / 256, 256, 0, stream>>>(sa, g2, AB_bf);
  gemm(pre_bf, KP, Wal_bf, KP, bal, nullptr, 0, P_bf, KP, NENT, E, EP16, KP);

  // ---- 9) fused big GEMM (LDS-staged via async copies) + sigmoid + path-loss ----
  {
    dim3 grid(NENT / 64, BB / 32);  // 625 x 32, exact tiling
    score_kernel<<<grid, 256, 0, stream>>>(AB_bf, P_bf, triples, score, accum);
  }

  // ---- 10) finalize losses ----
  finalize_kernel<<<1, 1, 0, stream>>>(accum, out);
}
